// DDSPGenerator_40450001994077
// MI455X (gfx1250) — matrix operand order repaired and checked by reading, not compiled, not run on hardware
//
#include <hip/hip_runtime.h>

typedef __attribute__((ext_vector_type(16))) _Float16 v16h;
typedef __attribute__((ext_vector_type(8)))  float    v8f;

#define BATCH   16
#define N_OSC   128
#define T0      64
#define UP_LEN  24576
#define FRAMES  1024
#define WINDOW  32
#define HOP     16
#define NBUF_LEN (HOP * (FRAMES - 1) + WINDOW)   // 16400
#define CROP    4096
#define OUT_T   16384
#define SRATE   11025.0f
#define NYQF    5512.5f
#define PADT    4                                 // zeroed time rows each side

// ---------------------------------------------------------------------------
// helpers: zero fill, x pack (transpose to [B,T,C] f16 + zero borders),
// weight repack [Cout,Cin,K] f32 -> [Cout,K,Cin] f16
// ---------------------------------------------------------------------------
__global__ void zero_f16_kernel(_Float16* __restrict__ p, int n) {
    int i = blockIdx.x * blockDim.x + threadIdx.x;
    if (i < n) p[i] = (_Float16)0.0f;
}

__global__ void zero_f32_kernel(float* __restrict__ p, int n) {
    int i = blockIdx.x * blockDim.x + threadIdx.x;
    if (i < n) p[i] = 0.0f;
}

__global__ void pack_x_kernel(const float* __restrict__ in, _Float16* __restrict__ out) {
    // in: [B,256,T0] -> out: [B, T0+2*PADT, 256] (borders zero)
    int idx = blockIdx.x * blockDim.x + threadIdx.x;
    const int total = BATCH * (T0 + 2 * PADT) * 256;
    if (idx >= total) return;
    int c   = idx % 256;
    int rem = idx / 256;
    int tp  = rem % (T0 + 2 * PADT);
    int b   = rem / (T0 + 2 * PADT);
    int t   = tp - PADT;
    float v = (t >= 0 && t < T0) ? in[((size_t)b * 256 + c) * T0 + t] : 0.0f;
    out[idx] = (_Float16)v;
}

__global__ void repack_w_kernel(const float* __restrict__ in, _Float16* __restrict__ out,
                                int Cout, int Cin, int K) {
    int idx = blockIdx.x * blockDim.x + threadIdx.x;
    if (idx >= Cout * Cin * K) return;
    int kk   = idx % K;
    int rem  = idx / K;
    int cin  = rem % Cin;
    int cout = rem / Cin;
    out[((size_t)cout * K + kk) * Cin + cin] = (_Float16)in[idx];
}

// ---------------------------------------------------------------------------
// WMMA conv-as-GEMM, time-major activations.
//   X  : f16 [B, Tsrc+2*PADT, Cin]   (borders zero -> no bounds checks)
//   W  : f16 [Cout(pad64), K, Cin]   (K-order kidx = kk*Cin + cin)
//   Out: MODE 0/2 -> f16 [B, Tout+2*PADT, Cout] (interior only)
//        MODE 1/3 -> f32 [B, Tout, Cout] (3: guarded to logical Cout, squared)
// Each wave: 64(M)x16(N) tile, 4 accumulators; per K-step: 8 A b128 + 2 B b128
// + 4 v_wmma, zero predication.  UPS folds nearest-x2 upsample into ts=tin>>1.
// ---------------------------------------------------------------------------
template<int KSIZE, int UPS, int MODE>
__global__ __launch_bounds__(256)
void wmma_conv_kernel(const _Float16* __restrict__ X, const _Float16* __restrict__ W,
                      const float* __restrict__ bias, void* __restrict__ Out,
                      int Cin, int Cout, int Tsrc, int pad)
{
    const int Tout  = UPS ? (Tsrc * 2) : Tsrc;
    const int Ncols = BATCH * Tout;

    const int lane  = threadIdx.x & 31;
    const int wave  = threadIdx.x >> 5;
    const int ntile = blockIdx.x * 8 + wave;
    if (ntile * 16 >= Ncols) return;
    const int m0 = blockIdx.y * 64;

    const int n   = lane & 15;
    const int hi  = lane >> 4;
    const int col = ntile * 16 + n;
    const int b   = col / Tout;
    const int t   = col - b * Tout;

    // fragment lane offsets (16-bit A 16x32 / B 32x16 layouts, ISA 7.12.2)
    const int offA0 = hi ? 8 : 0;
    const int offA1 = hi ? 24 : 16;
    const int offB  = hi ? 8 : 0;

    const int CinK = Cin * KSIZE;
    const _Float16* wrow[4];
    #pragma unroll
    for (int i = 0; i < 4; ++i)
        wrow[i] = W + (size_t)(m0 + i * 16 + (lane & 15)) * CinK;

    const _Float16* xb = X + (size_t)b * (Tsrc + 2 * PADT) * Cin;
    const int Cin32 = Cin >> 5;

    v8f acc0 = {}, acc1 = {}, acc2 = {}, acc3 = {};

    #pragma unroll
    for (int kk = 0; kk < KSIZE; ++kk) {
        const int tin = t + kk - pad;
        const int ts  = UPS ? (tin >> 1) : tin;        // borders land in zero pad
        const _Float16* brow = xb + (size_t)(ts + PADT) * Cin + offB;

        for (int cs = 0; cs < Cin32; ++cs) {
            const int kbase = kk * Cin + cs * 32;

            union { v16h h; uint4 u[2]; } Af[4];
            #pragma unroll
            for (int i = 0; i < 4; ++i) {
                Af[i].u[0] = *reinterpret_cast<const uint4*>(wrow[i] + kbase + offA0);
                Af[i].u[1] = *reinterpret_cast<const uint4*>(wrow[i] + kbase + offA1);
            }

            union { v16h h; uint4 u[2]; } Bf;
            const _Float16* bp = brow + cs * 32;
            Bf.u[0] = *reinterpret_cast<const uint4*>(bp);
            Bf.u[1] = *reinterpret_cast<const uint4*>(bp + 16);

            acc0 = __builtin_amdgcn_wmma_f32_16x16x32_f16(false, Af[0].h, false, Bf.h,
                                                          (short)0, acc0, false, false);
            acc1 = __builtin_amdgcn_wmma_f32_16x16x32_f16(false, Af[1].h, false, Bf.h,
                                                          (short)0, acc1, false, false);
            acc2 = __builtin_amdgcn_wmma_f32_16x16x32_f16(false, Af[2].h, false, Bf.h,
                                                          (short)0, acc2, false, false);
            acc3 = __builtin_amdgcn_wmma_f32_16x16x32_f16(false, Af[3].h, false, Bf.h,
                                                          (short)0, acc3, false, false);
        }
    }

    // epilogue: element v of tile i -> m = m0+i*16+v+8*hi (8 consecutive m)
    v8f accs[4] = {acc0, acc1, acc2, acc3};
    #pragma unroll
    for (int i = 0; i < 4; ++i) {
        const int mbase = m0 + i * 16 + (hi ? 8 : 0);
        if (MODE == 0 || MODE == 2) {
            union { _Float16 h[8]; uint4 u; } pk;
            #pragma unroll
            for (int v = 0; v < 8; ++v) {
                float c = accs[i][v];
                if (MODE == 2) c += bias[mbase + v];
                c = c > 0.0f ? c : 0.2f * c;
                pk.h[v] = (_Float16)c;
            }
            _Float16* op = (_Float16*)Out +
                ((size_t)b * (Tout + 2 * PADT) + (t + PADT)) * Cout + mbase;
            *reinterpret_cast<uint4*>(op) = pk.u;
        } else if (MODE == 1) {
            float* op = (float*)Out + ((size_t)b * Tout + t) * Cout + mbase;
            #pragma unroll
            for (int v = 0; v < 8; ++v) op[v] = accs[i][v];
        } else { // MODE 3: squared, guarded to logical Cout
            float* op = (float*)Out + ((size_t)b * Tout + t) * Cout;
            #pragma unroll
            for (int v = 0; v < 8; ++v) {
                int m = mbase + v;
                if (m < Cout) op[m] = accs[i][v] * accs[i][v];
            }
        }
    }
}

// ---------------------------------------------------------------------------
// osc head post: osc_f is time-major [B, T0, 256]
//   l = osc[:, :, 0:128]^2 ; f = center + tanh(osc[:, :, 128:])*0.5*erb
// outputs channel-major [B, 128, T0] for the oscillator bank
// ---------------------------------------------------------------------------
__global__ void osc_post_kernel(const float* __restrict__ osc,
                                float* __restrict__ lbuf, float* __restrict__ fbuf) {
    int idx = blockIdx.x * blockDim.x + threadIdx.x;
    if (idx >= BATCH * N_OSC * T0) return;
    int t  = idx & 63;
    int ch = (idx >> 6) & 127;
    int b  = idx >> 13;
    float center = 20.0f * powf(274.625f, (float)ch * (1.0f / 127.0f));
    float erb    = center * 0.108f + 24.7f;
    float lc = osc[((size_t)b * T0 + t) * 256 + ch];
    float fc = osc[((size_t)b * T0 + t) * 256 + 128 + ch];
    lbuf[idx] = lc * lc;
    fbuf[idx] = center + tanhf(fc) * 0.5f * erb;
}

// ---------------------------------------------------------------------------
// Oscillator bank: per (b,osc) row -> linear resize 64->24576, clamp freq,
// inclusive phase scan (LDS Hillis-Steele per 256 chunk + running carry),
// sin * amp accumulated into harmonic[b,t] via f32 global atomics.
// ---------------------------------------------------------------------------
__global__ __launch_bounds__(256)
void osc_bank_kernel(const float* __restrict__ fbuf, const float* __restrict__ lbuf,
                     float* __restrict__ harmonic) {
    __shared__ float sm[256];
    __shared__ float sf[T0];
    __shared__ float sl[T0];
    const int row = blockIdx.x;          // b*128 + osc
    const int b   = row >> 7;
    const int tid = threadIdx.x;
    if (tid < T0) {
        sf[tid] = fbuf[(size_t)row * T0 + tid];
        sl[tid] = lbuf[(size_t)row * T0 + tid];
    }
    __syncthreads();

    const float PH = 6.28318530718f / SRATE;
    float carry = 0.0f;
    for (int ch = 0; ch < UP_LEN / 256; ++ch) {
        int t = ch * 256 + tid;
        float u  = ((float)t + 0.5f) * (1.0f / 384.0f) - 0.5f;
        float fu = floorf(u);
        int   i0 = (int)fu;
        float fr = u - fu;
        int ia = i0     < 0 ? 0 : (i0     > 63 ? 63 : i0);
        int ib = i0 + 1 < 0 ? 0 : (i0 + 1 > 63 ? 63 : i0 + 1);
        float fv = sf[ia] + (sf[ib] - sf[ia]) * fr;
        float lv = sl[ia] + (sl[ib] - sl[ia]) * fr;
        fv = fminf(fmaxf(fv, 20.0f), NYQF);
        float df = fv * PH;

        sm[tid] = df;
        __syncthreads();
        float run = df;
        for (int off = 1; off < 256; off <<= 1) {
            float add = (tid >= off) ? sm[tid - off] : 0.0f;
            __syncthreads();
            run += add;
            sm[tid] = run;
            __syncthreads();
        }
        float phase = carry + run;
        float tot   = sm[255];
        atomicAdd(&harmonic[(size_t)b * UP_LEN + t], __sinf(phase) * lv);
        carry += tot;
        __syncthreads();
    }
}

// ---------------------------------------------------------------------------
// Noise bank: per (b,frame) 32-sample frame -> ortho rfft (17 bins) -> scale by
// n_l (time-major [B, FRAMES, 17]) -> ortho irfft -> overlap-add via atomics.
// ---------------------------------------------------------------------------
__global__ __launch_bounds__(32)
void noise_bank_kernel(const float* __restrict__ noise, const float* __restrict__ nl,
                       float* __restrict__ nbuf) {
    __shared__ float sn[WINDOW];
    __shared__ float sre[17];
    __shared__ float sim[17];
    const int fb    = blockIdx.x;        // b*1024 + frame
    const int tid   = threadIdx.x;       // 0..31
    const int b     = fb >> 10;
    const int frame = fb & 1023;

    sn[tid] = noise[(size_t)fb * WINDOW + tid] * 2.0f - 1.0f;
    __syncthreads();

    const float inv = 0.17677669529f;    // 1/sqrt(32)
    if (tid < 17) {
        float re = 0.0f, im = 0.0f;
        #pragma unroll
        for (int nn = 0; nn < WINDOW; ++nn) {
            float ang = -6.28318530718f * (float)(tid * nn) * (1.0f / 32.0f);
            float s, c;
            sincosf(ang, &s, &c);
            re += sn[nn] * c;
            im += sn[nn] * s;
        }
        float g = nl[(size_t)fb * 17 + tid] * inv;
        sre[tid] = re * g;
        sim[tid] = im * g;
    }
    __syncthreads();

    float acc = sre[0];
    #pragma unroll
    for (int k = 1; k < 16; ++k) {
        float ang = 6.28318530718f * (float)(k * tid) * (1.0f / 32.0f);
        float s, c;
        sincosf(ang, &s, &c);
        acc += 2.0f * (sre[k] * c - sim[k] * s);
    }
    acc += sre[16] * ((tid & 1) ? -1.0f : 1.0f);
    acc *= inv;

    atomicAdd(&nbuf[(size_t)b * NBUF_LEN + frame * HOP + tid], acc);
}

// ---------------------------------------------------------------------------
// Per-batch max |harmonic + padded-noise| over all 24576 samples, then crop/norm
// ---------------------------------------------------------------------------
__global__ __launch_bounds__(256)
void maxabs_kernel(const float* __restrict__ harmonic, const float* __restrict__ nbuf,
                   float* __restrict__ mx) {
    __shared__ float sm[256];
    const int b = blockIdx.x, tid = threadIdx.x;
    float m = 0.0f;
    for (int t = tid; t < UP_LEN; t += 256) {
        float v = harmonic[(size_t)b * UP_LEN + t];
        if (t >= CROP && t < CROP + OUT_T) v += nbuf[(size_t)b * NBUF_LEN + (t - CROP)];
        m = fmaxf(m, fabsf(v));
    }
    sm[tid] = m;
    __syncthreads();
    for (int off = 128; off > 0; off >>= 1) {
        if (tid < off) sm[tid] = fmaxf(sm[tid], sm[tid + off]);
        __syncthreads();
    }
    if (tid == 0) mx[b] = sm[0];
}

__global__ void finalize_kernel(const float* __restrict__ harmonic,
                                const float* __restrict__ nbuf,
                                const float* __restrict__ mx,
                                float* __restrict__ out) {
    int idx = blockIdx.x * blockDim.x + threadIdx.x;
    if (idx >= BATCH * OUT_T) return;
    int b = idx >> 14;
    int t = idx & (OUT_T - 1);
    float v = harmonic[(size_t)b * UP_LEN + CROP + t] + nbuf[(size_t)b * NBUF_LEN + t];
    out[idx] = v / (mx[b] + 1e-8f);
}

// ---------------------------------------------------------------------------
// Host launcher
// ---------------------------------------------------------------------------
template<int KSIZE, int UPS, int MODE>
static void launch_gemm(const _Float16* X, const _Float16* W, const float* bias, void* Out,
                        int Cin, int Cout, int Tsrc, int pad, hipStream_t stream) {
    int Tout  = UPS ? Tsrc * 2 : Tsrc;
    int Ncols = BATCH * Tout;
    dim3 grid((Ncols + 127) / 128, (Cout + 63) / 64);
    wmma_conv_kernel<KSIZE, UPS, MODE><<<grid, 256, 0, stream>>>(X, W, bias, Out,
                                                                 Cin, Cout, Tsrc, pad);
}

extern "C" void kernel_launch(void* const* d_in, const int* in_sizes, int n_in,
                              void* d_out, int out_size, void* d_ws, size_t ws_size,
                              hipStream_t stream) {
    const float* x        = (const float*)d_in[0];
    const float* w_main[4]= {(const float*)d_in[1], (const float*)d_in[2],
                             (const float*)d_in[3], (const float*)d_in[4]};
    const float* w_freq   = (const float*)d_in[5];
    const float* w_nl[4]  = {(const float*)d_in[6], (const float*)d_in[8],
                             (const float*)d_in[10], (const float*)d_in[12]};
    const float* b_nl[4]  = {(const float*)d_in[7], (const float*)d_in[9],
                             (const float*)d_in[11], (const float*)d_in[13]};
    const float* w_nloud  = (const float*)d_in[14];
    const float* noise    = (const float*)d_in[15];

    // ---- workspace carve-out (256B aligned) ----
    char* base = (char*)d_ws;
    size_t off = 0;
    auto alloc = [&](size_t bytes) -> void* {
        void* p = base + off;
        off = (off + bytes + 255) & ~(size_t)255;
        return p;
    };
    auto padrows = [](int T) { return T + 2 * PADT; };

    const int nwm0    = 512 * 256;
    const int nwm     = 512 * 512 * 3;
    const int nwf     = 256 * 512;
    const int nwnl    = 512 * 512 * 7;
    const int nwnloud_pad = 64 * 512;             // Cout 17 padded to 64 rows

    _Float16* x_t    = (_Float16*)alloc((size_t)BATCH * padrows(T0) * 256 * 2);
    _Float16* wm0_h  = (_Float16*)alloc((size_t)nwm0 * 2);
    _Float16* wm_h[3];
    for (int i = 0; i < 3; ++i) wm_h[i] = (_Float16*)alloc((size_t)nwm * 2);
    _Float16* wf_h   = (_Float16*)alloc((size_t)nwf * 2);
    _Float16* wnl_h[4];
    for (int i = 0; i < 4; ++i) wnl_h[i] = (_Float16*)alloc((size_t)nwnl * 2);
    _Float16* wnloud_h = (_Float16*)alloc((size_t)nwnloud_pad * 2);

    const int nh    = BATCH * padrows(T0) * 512;
    _Float16* h_a   = (_Float16*)alloc((size_t)nh * 2);
    _Float16* h_b   = (_Float16*)alloc((size_t)nh * 2);
    const int ny128  = BATCH * padrows(128)  * 512;
    const int ny256  = BATCH * padrows(256)  * 512;
    const int ny512  = BATCH * padrows(512)  * 512;
    const int ny1024 = BATCH * padrows(1024) * 512;
    _Float16* y128  = (_Float16*)alloc((size_t)ny128  * 2);
    _Float16* y256  = (_Float16*)alloc((size_t)ny256  * 2);
    _Float16* y512  = (_Float16*)alloc((size_t)ny512  * 2);
    _Float16* y1024 = (_Float16*)alloc((size_t)ny1024 * 2);

    float* osc_f    = (float*)alloc((size_t)BATCH * T0 * 256 * 4);      // [B,T,256]
    float* l_buf    = (float*)alloc((size_t)BATCH * N_OSC * T0 * 4);    // [B,128,T]
    float* f_buf    = (float*)alloc((size_t)BATCH * N_OSC * T0 * 4);
    float* n_l      = (float*)alloc((size_t)BATCH * FRAMES * 17 * 4);   // [B,T,17]
    float* harmonic = (float*)alloc((size_t)BATCH * UP_LEN * 4);
    float* nbuf     = (float*)alloc((size_t)BATCH * NBUF_LEN * 4);
    float* mx       = (float*)alloc(64 * 4);

    // ---- pack input, repack weights, zero activation buffers (borders!) ----
    {
        int n = BATCH * padrows(T0) * 256;
        pack_x_kernel<<<(n + 255) / 256, 256, 0, stream>>>(x, x_t);
    }
    auto repack = [&](const float* src, _Float16* dst, int Cout, int Cin, int K) {
        int n = Cout * Cin * K;
        repack_w_kernel<<<(n + 255) / 256, 256, 0, stream>>>(src, dst, Cout, Cin, K);
    };
    auto zf16 = [&](_Float16* p, int n) {
        zero_f16_kernel<<<(n + 255) / 256, 256, 0, stream>>>(p, n);
    };
    zf16(wnloud_h, nwnloud_pad);                  // zero padded head rows first
    repack(w_main[0], wm0_h, 512, 256, 1);
    for (int i = 0; i < 3; ++i) repack(w_main[i + 1], wm_h[i], 512, 512, 3);
    repack(w_freq, wf_h, 256, 512, 1);
    for (int i = 0; i < 4; ++i) repack(w_nl[i], wnl_h[i], 512, 512, 7);
    repack(w_nloud, wnloud_h, 17, 512, 1);

    zf16(h_a, nh);  zf16(h_b, nh);
    zf16(y128, ny128);  zf16(y256, ny256);  zf16(y512, ny512);  zf16(y1024, ny1024);
    zero_f32_kernel<<<(BATCH * UP_LEN + 255) / 256, 256, 0, stream>>>(harmonic, BATCH * UP_LEN);
    zero_f32_kernel<<<(BATCH * NBUF_LEN + 255) / 256, 256, 0, stream>>>(nbuf, BATCH * NBUF_LEN);

    // ---- main stack (lrelu 0.2, bias-free) ----
    launch_gemm<1, 0, 0>(x_t, wm0_h,   nullptr, h_a, 256, 512, T0, 0, stream);
    launch_gemm<3, 0, 0>(h_a, wm_h[0], nullptr, h_b, 512, 512, T0, 1, stream);
    launch_gemm<3, 0, 0>(h_b, wm_h[1], nullptr, h_a, 512, 512, T0, 1, stream);
    launch_gemm<3, 0, 0>(h_a, wm_h[2], nullptr, h_b, 512, 512, T0, 1, stream);  // h = h_b

    // ---- oscillator head ----
    launch_gemm<1, 0, 1>(h_b, wf_h, nullptr, osc_f, 512, 256, T0, 0, stream);
    osc_post_kernel<<<(BATCH * N_OSC * T0 + 255) / 256, 256, 0, stream>>>(osc_f, l_buf, f_buf);

    // ---- noise-loudness branch (upsample2 folded into gather, conv k7) ----
    launch_gemm<7, 1, 2>(h_b,  wnl_h[0], b_nl[0], y128,  512, 512, 64,  3, stream);
    launch_gemm<7, 1, 2>(y128, wnl_h[1], b_nl[1], y256,  512, 512, 128, 3, stream);
    launch_gemm<7, 1, 2>(y256, wnl_h[2], b_nl[2], y512,  512, 512, 256, 3, stream);
    launch_gemm<7, 1, 2>(y512, wnl_h[3], b_nl[3], y1024, 512, 512, 512, 3, stream);
    launch_gemm<1, 0, 3>(y1024, wnloud_h, nullptr, n_l,  512, 17, 1024, 0, stream);

    // ---- synthesis ----
    osc_bank_kernel<<<BATCH * N_OSC, 256, 0, stream>>>(f_buf, l_buf, harmonic);
    noise_bank_kernel<<<BATCH * FRAMES, 32, 0, stream>>>(noise, n_l, nbuf);
    maxabs_kernel<<<BATCH, 256, 0, stream>>>(harmonic, nbuf, mx);
    finalize_kernel<<<(BATCH * OUT_T + 255) / 256, 256, 0, stream>>>(harmonic, nbuf, mx,
                                                                     (float*)d_out);
    (void)in_sizes; (void)n_in; (void)out_size; (void)ws_size;
}
